// EncoderDecoder_56401510531105
// MI455X (gfx1250) — matrix operand order repaired and checked by reading, not compiled, not run on hardware
//
#include <hip/hip_runtime.h>

typedef __bf16 bf16;
typedef __attribute__((ext_vector_type(16))) __bf16 bf16x16;
typedef __attribute__((ext_vector_type(8)))  __bf16 bf16x8;
typedef __attribute__((ext_vector_type(8)))  float  f32x8;

#define DEVINL __device__ __forceinline__

constexpr int B = 32, T = 512, F = 64, H = 8, U = 128, OUT_STEPS = 64;
constexpr int HU = H * U;   // 1024
constexpr int G4 = 4 * U;   // 512 (i,f,g,o gates)
constexpr float SCORE_SCALE = 0.08838834764831845f; // 1/sqrt(U)

DEVINL int lane_id() { return (int)(threadIdx.x & 31); }

// A fragment: 16x32 bf16, row-major at `base` with leading dim `ld` (elements).
// Layout: lane(0-15)=rows M=0..15, K = hi*16 .. hi*16+15 contiguous (matches frag_bt),
// where hi = lane>>4.
DEVINL bf16x16 frag_a(const bf16* base, int ld) {
  const int lane = lane_id();
  const bf16* r = base + (size_t)(lane & 15) * ld + (lane >> 4) * 16;
  bf16x8 a0 = *(const bf16x8*)(r);
  bf16x8 a1 = *(const bf16x8*)(r + 8);
  bf16x16 a;
#pragma unroll
  for (int i = 0; i < 8; ++i) { a[i] = a0[i]; a[i + 8] = a1[i]; }
  return a;
}

// B fragment via B^T stored row-major [N][K]: lane holds column n=lane&15,
// K = (lane>>4)*16 .. +15 contiguous -> one 32-byte load.
DEVINL bf16x16 frag_bt(const bf16* baseT, int ld) {
  const int lane = lane_id();
  return *(const bf16x16*)(baseT + (size_t)(lane & 15) * ld + (lane >> 4) * 16);
}

DEVINL f32x8 wmma_bf16(bf16x16 a, bf16x16 b, f32x8 c) {
  return __builtin_amdgcn_wmma_f32_16x16x32_bf16(false, a, false, b, (short)0, c,
                                                 false, false);
}

template <int W>
DEVINL float red_max(float v) {
#pragma unroll
  for (int m = W / 2; m >= 1; m >>= 1) v = fmaxf(v, __shfl_xor(v, m, W));
  return v;
}
template <int W>
DEVINL float red_sum(float v) {
#pragma unroll
  for (int m = W / 2; m >= 1; m >>= 1) v += __shfl_xor(v, m, W);
  return v;
}
DEVINL float sigm(float x) { return 1.0f / (1.0f + __expf(-x)); }

// ---------------------------------------------------------------- conversions
__global__ void cvt_k(const float* __restrict__ in, bf16* __restrict__ out, int n) {
  int i = blockIdx.x * 256 + threadIdx.x;
  if (i < n) out[i] = (bf16)in[i];
}

// out[c][r] = (bf16) in[r][c]   (in: rows x cols, f32)
__global__ void transpose_cvt_k(const float* __restrict__ in, bf16* __restrict__ out,
                                int rows, int cols) {
  int i = blockIdx.x * 256 + threadIdx.x;
  if (i >= rows * cols) return;
  int r = i / cols, c = i % cols;
  out[(size_t)c * rows + r] = (bf16)in[i];
}

// ------------------------------------------------------------- generic GEMMs
// C[M][N] = A[M][K] * B[K][N], BT stored [N][K]; one 16x16 tile per wave.
__global__ void gemm_bt_bf16_k(const bf16* __restrict__ A, const bf16* __restrict__ BT,
                               bf16* __restrict__ C, int M, int N, int K) {
  const int wid = threadIdx.x >> 5;
  const int tiles_n = N >> 4;
  int tile = blockIdx.x * 8 + wid;
  if (tile >= (M >> 4) * tiles_n) return;
  int mt = tile / tiles_n, nt = tile % tiles_n;
  const bf16* a = A + (size_t)mt * 16 * K;
  const bf16* b = BT + (size_t)nt * 16 * K;
  f32x8 acc = {};
  for (int k0 = 0; k0 < K; k0 += 32) {
    __builtin_prefetch(a + k0 + 64, 0, 0);
    acc = wmma_bf16(frag_a(a + k0, K), frag_bt(b + k0, K), acc);
  }
  const int lo = lane_id() & 15, hi = lane_id() >> 4;
#pragma unroll
  for (int j = 0; j < 8; ++j)
    C[(size_t)(mt * 16 + hi * 8 + j) * N + nt * 16 + lo] = (bf16)acc[j];
}

__global__ void gemm_bt_f32_bias_k(const bf16* __restrict__ A, const bf16* __restrict__ BT,
                                   const float* __restrict__ bias, float* __restrict__ C,
                                   int M, int N, int K) {
  const int wid = threadIdx.x >> 5;
  const int tiles_n = N >> 4;
  int tile = blockIdx.x * 8 + wid;
  if (tile >= (M >> 4) * tiles_n) return;
  int mt = tile / tiles_n, nt = tile % tiles_n;
  const bf16* a = A + (size_t)mt * 16 * K;
  const bf16* b = BT + (size_t)nt * 16 * K;
  f32x8 acc = {};
  for (int k0 = 0; k0 < K; k0 += 32)
    acc = wmma_bf16(frag_a(a + k0, K), frag_bt(b + k0, K), acc);
  const int lo = lane_id() & 15, hi = lane_id() >> 4;
  float bi = bias[nt * 16 + lo];
#pragma unroll
  for (int j = 0; j < 8; ++j)
    C[(size_t)(mt * 16 + hi * 8 + j) * N + nt * 16 + lo] = acc[j] + bi;
}

// --------------------------------------------------------------- QKV project
// xb [B*T][F] bf16, wT [H*U][F] bf16 -> out [B,H,T,U]; transposed=1 -> [B,H,U,T].
__global__ void proj_qkv_k(const bf16* __restrict__ xb, const bf16* __restrict__ wT,
                           bf16* __restrict__ out, int transposed) {
  const int wid = threadIdx.x >> 5;
  int tile = blockIdx.x * 8 + wid;          // B*H*(T/16)*(U/16) = 65536 tiles
  int bh = tile >> 8;                       // 256 tiles per (b,h)
  int r = tile & 255;
  int mt = r >> 3, nt = r & 7;              // mt: t-tile (0..31), nt: u-tile (0..7)
  int b = bh >> 3, h = bh & 7;
  const bf16* a = xb + ((size_t)b * T + mt * 16) * F;
  const bf16* w = wT + ((size_t)h * U + nt * 16) * F;
  f32x8 acc = {};
#pragma unroll
  for (int k0 = 0; k0 < F; k0 += 32)
    acc = wmma_bf16(frag_a(a + k0, F), frag_bt(w + k0, F), acc);
  const int lo = lane_id() & 15, hi = lane_id() >> 4;
  if (!transposed) {
    bf16* o = out + ((size_t)bh * T + mt * 16) * U + nt * 16;
#pragma unroll
    for (int j = 0; j < 8; ++j) o[(size_t)(hi * 8 + j) * U + lo] = (bf16)acc[j];
  } else {
    bf16* o = out + ((size_t)bh * U + nt * 16) * T + mt * 16;
#pragma unroll
    for (int j = 0; j < 8; ++j) o[(size_t)lo * T + hi * 8 + j] = (bf16)acc[j];
  }
}

// ------------------------------------------------------------------ attention
// Per block: one (b,h) and 64 query rows. S=Q*K^T (LDS f32), softmax, ctx=P*V.
// ctx stored permuted to [b][t][h][u] bf16 for the output projection GEMM.
__global__ void attention_k(const bf16* __restrict__ Q, const bf16* __restrict__ Km,
                            const bf16* __restrict__ VT, bf16* __restrict__ ctx) {
  extern __shared__ char smem[];
  float* sS = (float*)smem;                               // [64][512] 128KB
  bf16* sP = (bf16*)(smem + 64 * 512 * 4);                // [64][512]  64KB
  bf16* sQ = (bf16*)(smem + 64 * 512 * 4 + 64 * 512 * 2); // [64][128]  16KB
  const int tb = blockIdx.x, h = blockIdx.y, b = blockIdx.z;
  const int bh = b * H + h, t0 = tb * 64;
  const int tid = threadIdx.x, wid = tid >> 5, lane = tid & 31;
  const int lo = lane & 15, hi = lane >> 4;

  const bf16* qg = Q + ((size_t)bh * T + t0) * U;
  for (int i = tid; i < 64 * U; i += 256) sQ[i] = qg[i];
  __syncthreads();

  // scores: 4 m-tiles x 32 n-tiles
  for (int tile = wid; tile < 128; tile += 8) {
    int mt = tile >> 5, nt = tile & 31;
    const bf16* kb = Km + ((size_t)bh * T + nt * 16) * U;
    f32x8 acc = {};
#pragma unroll
    for (int k0 = 0; k0 < U; k0 += 32)
      acc = wmma_bf16(frag_a(sQ + mt * 16 * U + k0, U), frag_bt(kb + k0, U), acc);
#pragma unroll
    for (int j = 0; j < 8; ++j)
      sS[(mt * 16 + hi * 8 + j) * 512 + nt * 16 + lo] = acc[j] * SCORE_SCALE;
  }
  __syncthreads();

  // softmax: each wave owns 8 rows; 32 lanes x 16 elements per row
  for (int rr = 0; rr < 8; ++rr) {
    int r = wid * 8 + rr;
    const float* srow = sS + r * 512;
    float vals[16], mx = -3.0e38f;
#pragma unroll
    for (int i = 0; i < 16; ++i) { vals[i] = srow[lane + 32 * i]; mx = fmaxf(mx, vals[i]); }
    mx = red_max<32>(mx);
    float sum = 0.f;
#pragma unroll
    for (int i = 0; i < 16; ++i) { vals[i] = __expf(vals[i] - mx); sum += vals[i]; }
    sum = red_sum<32>(sum);
    float inv = 1.0f / sum;
    bf16* prow = sP + r * 512;
#pragma unroll
    for (int i = 0; i < 16; ++i) prow[lane + 32 * i] = (bf16)(vals[i] * inv);
  }
  __syncthreads();

  // ctx = P * V : 4 m-tiles x 8 n-tiles, K=512
  for (int tile = wid; tile < 32; tile += 8) {
    int mt = tile >> 3, nt = tile & 7;
    const bf16* vb = VT + ((size_t)bh * U + nt * 16) * T;
    f32x8 acc = {};
#pragma unroll
    for (int k0 = 0; k0 < T; k0 += 32)
      acc = wmma_bf16(frag_a(sP + mt * 16 * T + k0, T), frag_bt(vb + k0, T), acc);
#pragma unroll
    for (int j = 0; j < 8; ++j) {
      int t = t0 + mt * 16 + hi * 8 + j;
      ctx[((size_t)(b * T + t) * H + h) * U + nt * 16 + lo] = (bf16)acc[j];
    }
  }
}

// -------------------------------------------------------------- encoder LSTM
// 2 blocks x 16 batch rows. recT [512][128] bf16 resident in LDS. 512 steps.
__global__ void lstm_enc_k(const float* __restrict__ xz, const bf16* __restrict__ recT,
                           bf16* __restrict__ hlast) {
  extern __shared__ char smem[];
  bf16* sRec = (bf16*)smem;                                // [512][128] 128KB
  float* sZ = (float*)(smem + G4 * U * 2);                 // [16][512]   32KB
  bf16* sH = (bf16*)(smem + G4 * U * 2 + 16 * G4 * 4);     // [16][128]    4KB
  const int b0 = blockIdx.x * 16;
  const int tid = threadIdx.x, wid = tid >> 5, lane = tid & 31;
  const int lo = lane & 15, hi = lane >> 4;
  for (int i = tid; i < G4 * U; i += 256) sRec[i] = recT[i];
  for (int i = tid; i < 16 * U; i += 256) sH[i] = (bf16)0.0f;
  __syncthreads();

  const int row = tid >> 4, sub = tid & 15; // pointwise: row 0..15, 8 u's each
  float c[8];
#pragma unroll
  for (int j = 0; j < 8; ++j) c[j] = 0.f;

  for (int t = 0; t < T; ++t) {
    // z = xz[:,t,:] + h @ rec  : 1x32 tiles, 4 per wave
#pragma unroll
    for (int i = 0; i < 4; ++i) {
      int nt = wid * 4 + i;
      f32x8 acc = {};
#pragma unroll
      for (int k0 = 0; k0 < U; k0 += 32)
        acc = wmma_bf16(frag_a(sH + k0, U), frag_bt(sRec + (size_t)nt * 16 * U + k0, U), acc);
#pragma unroll
      for (int j = 0; j < 8; ++j) {
        int m = hi * 8 + j, n = nt * 16 + lo;
        sZ[m * G4 + n] = acc[j] + xz[((size_t)(b0 + m) * T + t) * G4 + n];
      }
    }
    __syncthreads();

    const float* zr = sZ + row * G4;
    float gv[8], gmx = -3.0e38f;
#pragma unroll
    for (int j = 0; j < 8; ++j) { gv[j] = zr[2 * U + sub * 8 + j]; gmx = fmaxf(gmx, gv[j]); }
    gmx = red_max<16>(gmx);
    float gsum = 0.f;
#pragma unroll
    for (int j = 0; j < 8; ++j) { gv[j] = __expf(gv[j] - gmx); gsum += gv[j]; }
    gsum = red_sum<16>(gsum);
    float ginv = 1.0f / gsum;

    float ov[8], ev[8], cmx = -3.0e38f, csum = 0.f;
#pragma unroll
    for (int j = 0; j < 8; ++j) {
      float iv = sigm(zr[sub * 8 + j]);
      float fv = sigm(zr[U + sub * 8 + j]);
      ov[j] = sigm(zr[3 * U + sub * 8 + j]);
      c[j] = fv * c[j] + iv * (gv[j] * ginv);
      cmx = fmaxf(cmx, c[j]);
    }
    cmx = red_max<16>(cmx);
#pragma unroll
    for (int j = 0; j < 8; ++j) { ev[j] = __expf(c[j] - cmx); csum += ev[j]; }
    csum = red_sum<16>(csum);
    float cinv = 1.0f / csum;
#pragma unroll
    for (int j = 0; j < 8; ++j) sH[row * U + sub * 8 + j] = (bf16)(ov[j] * ev[j] * cinv);
    __syncthreads();
  }
  for (int i = tid; i < 16 * U; i += 256) hlast[(size_t)b0 * U + i] = sH[i];
}

// -------------------------------------------------------------- decoder LSTM
// Input is h_last repeated => z_in computed once. 64 steps, writes hseq bf16.
__global__ void lstm_dec_k(const bf16* __restrict__ hlast, const bf16* __restrict__ dkT,
                           const float* __restrict__ dbias, const bf16* __restrict__ recT,
                           bf16* __restrict__ hseq) {
  extern __shared__ char smem[];
  bf16* sRec = (bf16*)smem;                                   // [512][128]
  float* sZin = (float*)(smem + G4 * U * 2);                  // [16][512]
  float* sZ = (float*)(smem + G4 * U * 2 + 16 * G4 * 4);      // [16][512]
  bf16* sH = (bf16*)(smem + G4 * U * 2 + 2 * 16 * G4 * 4);    // [16][128]
  const int b0 = blockIdx.x * 16;
  const int tid = threadIdx.x, wid = tid >> 5, lane = tid & 31;
  const int lo = lane & 15, hi = lane >> 4;
  for (int i = tid; i < G4 * U; i += 256) sRec[i] = recT[i];
  for (int i = tid; i < 16 * U; i += 256) sH[i] = (bf16)0.0f;

  // z_in = h_last @ dec_kernel + bias (same for every step)
#pragma unroll
  for (int i = 0; i < 4; ++i) {
    int nt = wid * 4 + i;
    f32x8 acc = {};
#pragma unroll
    for (int k0 = 0; k0 < U; k0 += 32)
      acc = wmma_bf16(frag_a(hlast + (size_t)b0 * U + k0, U),
                      frag_bt(dkT + (size_t)nt * 16 * U + k0, U), acc);
#pragma unroll
    for (int j = 0; j < 8; ++j) {
      int m = hi * 8 + j, n = nt * 16 + lo;
      sZin[m * G4 + n] = acc[j] + dbias[n];
    }
  }
  __syncthreads();

  const int row = tid >> 4, sub = tid & 15;
  float c[8];
#pragma unroll
  for (int j = 0; j < 8; ++j) c[j] = 0.f;

  for (int t = 0; t < OUT_STEPS; ++t) {
#pragma unroll
    for (int i = 0; i < 4; ++i) {
      int nt = wid * 4 + i;
      f32x8 acc = {};
#pragma unroll
      for (int k0 = 0; k0 < U; k0 += 32)
        acc = wmma_bf16(frag_a(sH + k0, U), frag_bt(sRec + (size_t)nt * 16 * U + k0, U), acc);
#pragma unroll
      for (int j = 0; j < 8; ++j) {
        int m = hi * 8 + j, n = nt * 16 + lo;
        sZ[m * G4 + n] = acc[j] + sZin[m * G4 + n];
      }
    }
    __syncthreads();

    const float* zr = sZ + row * G4;
    float gv[8], gmx = -3.0e38f;
#pragma unroll
    for (int j = 0; j < 8; ++j) { gv[j] = zr[2 * U + sub * 8 + j]; gmx = fmaxf(gmx, gv[j]); }
    gmx = red_max<16>(gmx);
    float gsum = 0.f;
#pragma unroll
    for (int j = 0; j < 8; ++j) { gv[j] = __expf(gv[j] - gmx); gsum += gv[j]; }
    gsum = red_sum<16>(gsum);
    float ginv = 1.0f / gsum;

    float ov[8], ev[8], cmx = -3.0e38f, csum = 0.f;
#pragma unroll
    for (int j = 0; j < 8; ++j) {
      float iv = sigm(zr[sub * 8 + j]);
      float fv = sigm(zr[U + sub * 8 + j]);
      ov[j] = sigm(zr[3 * U + sub * 8 + j]);
      c[j] = fv * c[j] + iv * (gv[j] * ginv);
      cmx = fmaxf(cmx, c[j]);
    }
    cmx = red_max<16>(cmx);
#pragma unroll
    for (int j = 0; j < 8; ++j) { ev[j] = __expf(c[j] - cmx); csum += ev[j]; }
    csum = red_sum<16>(csum);
    float cinv = 1.0f / csum;
#pragma unroll
    for (int j = 0; j < 8; ++j) {
      float hv = ov[j] * ev[j] * cinv;
      sH[row * U + sub * 8 + j] = (bf16)hv;
      hseq[((size_t)(b0 + row) * OUT_STEPS + t) * U + sub * 8 + j] = (bf16)hv;
    }
    __syncthreads();
  }
}

// --------------------------------------------------------------------- host
extern "C" void kernel_launch(void* const* d_in, const int* in_sizes, int n_in,
                              void* d_out, int out_size, void* d_ws, size_t ws_size,
                              hipStream_t stream) {
  (void)in_sizes; (void)n_in; (void)out_size; (void)ws_size;
  const float* x = (const float*)d_in[0];
  const float* wq = (const float*)d_in[1];
  const float* wk = (const float*)d_in[2];
  const float* wv = (const float*)d_in[3];
  const float* wo = (const float*)d_in[4];
  const float* ek = (const float*)d_in[5];
  const float* er = (const float*)d_in[6];
  const float* eb = (const float*)d_in[7];
  const float* dk = (const float*)d_in[8];
  const float* dr = (const float*)d_in[9];
  const float* db = (const float*)d_in[10];
  const float* dw = (const float*)d_in[11];
  const float* dnb = (const float*)d_in[12];
  float* out = (float*)d_out;

  char* ws = (char*)d_ws;
  size_t off = 0;
  auto take = [&](size_t bytes) -> char* {
    char* p = ws + off;
    off = (off + bytes + 255) & ~(size_t)255;
    return p;
  };
  bf16* xb = (bf16*)take((size_t)B * T * F * 2);
  bf16* wqT = (bf16*)take((size_t)HU * F * 2);
  bf16* wkT = (bf16*)take((size_t)HU * F * 2);
  bf16* wvT = (bf16*)take((size_t)HU * F * 2);
  bf16* woT = (bf16*)take((size_t)F * HU * 2);
  bf16* ekT = (bf16*)take((size_t)G4 * F * 2);
  bf16* erT = (bf16*)take((size_t)G4 * U * 2);
  bf16* dkT = (bf16*)take((size_t)G4 * U * 2);
  bf16* drT = (bf16*)take((size_t)G4 * U * 2);
  bf16* dwT = (bf16*)take((size_t)F * U * 2);
  bf16* Qg = (bf16*)take((size_t)B * H * T * U * 2);
  bf16* Kg = (bf16*)take((size_t)B * H * T * U * 2);
  bf16* VTg = (bf16*)take((size_t)B * H * T * U * 2);
  bf16* ctx = (bf16*)take((size_t)B * T * HU * 2);
  bf16* hatt = (bf16*)take((size_t)B * T * F * 2);
  bf16* hlast = (bf16*)take((size_t)B * U * 2);
  bf16* hseq = (bf16*)take((size_t)B * OUT_STEPS * U * 2);
  float* xz = (float*)Qg; // alias: Q dead after attention; sizes match (32MB)

  cvt_k<<<(B * T * F) / 256, 256, 0, stream>>>(x, xb, B * T * F);
  transpose_cvt_k<<<(F * HU) / 256, 256, 0, stream>>>(wq, wqT, F, HU);
  transpose_cvt_k<<<(F * HU) / 256, 256, 0, stream>>>(wk, wkT, F, HU);
  transpose_cvt_k<<<(F * HU) / 256, 256, 0, stream>>>(wv, wvT, F, HU);
  transpose_cvt_k<<<(HU * F) / 256, 256, 0, stream>>>(wo, woT, HU, F);
  transpose_cvt_k<<<(F * G4) / 256, 256, 0, stream>>>(ek, ekT, F, G4);
  transpose_cvt_k<<<(U * G4) / 256, 256, 0, stream>>>(er, erT, U, G4);
  transpose_cvt_k<<<(U * G4) / 256, 256, 0, stream>>>(dk, dkT, U, G4);
  transpose_cvt_k<<<(U * G4) / 256, 256, 0, stream>>>(dr, drT, U, G4);
  transpose_cvt_k<<<(U * F + 255) / 256, 256, 0, stream>>>(dw, dwT, U, F);

  proj_qkv_k<<<8192, 256, 0, stream>>>(xb, wqT, Qg, 0);
  proj_qkv_k<<<8192, 256, 0, stream>>>(xb, wkT, Kg, 0);
  proj_qkv_k<<<8192, 256, 0, stream>>>(xb, wvT, VTg, 1);

  attention_k<<<dim3(T / 64, H, B), 256, 212992, stream>>>(Qg, Kg, VTg, ctx);

  // ctx [B*T][HU] x woT -> h_attn bf16
  gemm_bt_bf16_k<<<512, 256, 0, stream>>>(ctx, woT, hatt, B * T, F, HU);
  // encoder input projection (+bias) -> xz f32 (aliases Qg)
  gemm_bt_f32_bias_k<<<4096, 256, 0, stream>>>(hatt, ekT, eb, xz, B * T, G4, F);

  lstm_enc_k<<<2, 256, 167936, stream>>>(xz, erT, hlast);
  lstm_dec_k<<<2, 256, 200704, stream>>>(hlast, dkT, db, drT, hseq);

  // final TimeDistributed Dense -> d_out f32
  gemm_bt_f32_bias_k<<<64, 256, 0, stream>>>(hseq, dwT, dnb, out, B * OUT_STEPS, F, U);
}
